// Spatial_Attention_layer_40286793236889
// MI455X (gfx1250) — matrix operand order repaired and checked by reading, not compiled, not run on hardware
//
#include <hip/hip_runtime.h>
#include <hip/hip_bf16.h>
#include <math.h>

// B=16, N=2048, C_IN=C_OUT=64
#define BB 16
#define NN 2048
#define CC 64

typedef __attribute__((ext_vector_type(16))) _Float16 v16h;
typedef __attribute__((ext_vector_type(8)))  _Float16 v8h;
typedef __attribute__((ext_vector_type(8)))  float    v8f;

__device__ __forceinline__ v16h make_a(v8h lo, v8h hi) {
  v16h a;
#pragma unroll
  for (int i = 0; i < 8; ++i) { a[i] = lo[i]; a[i + 8] = hi[i]; }
  return a;
}

// A operand, 16(M) x 32(K) f16, rows row0..row0+15 of row-major src (ld elems/row).
// ISA layout: lane<16 -> M=lane, elems0-7=K0..7, elems8-15=K16..23;
//             lane>=16 -> M=lane-16, elems0-7=K8..15, elems8-15=K24..31.
__device__ __forceinline__ v16h load_a_tile(const _Float16* src, int ld, int row0, int k0, int lane) {
  const int m = lane & 15, h = (lane >> 4) & 1;
  const _Float16* p = src + (size_t)(row0 + m) * ld + k0 + h * 8;
  return make_a(*(const v8h*)p, *(const v8h*)(p + 16));
}

// B operand, 32(K) x 16(N) f16 where column n of B = row (row0+n) of row-major src,
// K runs along the row. ISA layout: lanes 0-15 hold K0..7,K16..23 of column lane,
// lanes 16-31 hold K8..15,K24..31 of column lane-16.
__device__ __forceinline__ v16h load_b_tile(const _Float16* src, int ld, int row0, int k0, int lane) {
  const int n = lane & 15, h = (lane >> 4) & 1;
  const _Float16* p = src + (size_t)(row0 + n) * ld + k0 + h * 8;
  return make_a(*(const v8h*)p, *(const v8h*)(p + 16));
}

__device__ __forceinline__ v8f wmma32(v16h a, v16h b, v8f c) {
  return __builtin_amdgcn_wmma_f32_16x16x32_f16(false, a, false, b, (short)0, c, false, false);
}

// ---------------- Kernel 1: QKV projection (tiny: 0.8 GFLOP) ------------------
// Q = x@Wq^T + bq (f16), K = x@Wk^T + bk (f16), V stored TRANSPOSED: Vt[b][c][n].
__global__ void k_qkv(const float* __restrict__ x,  const float* __restrict__ Wq,
                      const float* __restrict__ bq, const float* __restrict__ Wk,
                      const float* __restrict__ bk, const float* __restrict__ Wv,
                      _Float16* __restrict__ Qh, _Float16* __restrict__ Kh,
                      _Float16* __restrict__ Vt) {
  __shared__ float xs[CC];
  const int row = blockIdx.x;             // b*N + n
  const int b = row / NN, n = row % NN;
  const int t = threadIdx.x;              // output channel 0..63
  xs[t] = x[(size_t)row * CC + t];
  __syncthreads();
  float aq = bq[t], ak = bk[t], av = 0.f;
#pragma unroll 8
  for (int k = 0; k < CC; ++k) {
    const float xv = xs[k];
    aq = fmaf(xv, Wq[t * CC + k], aq);
    ak = fmaf(xv, Wk[t * CC + k], ak);
    av = fmaf(xv, Wv[t * CC + k], av);
  }
  Qh[(size_t)row * CC + t] = (_Float16)aq;
  Kh[(size_t)row * CC + t] = (_Float16)ak;
  Vt[((size_t)b * CC + t) * NN + n] = (_Float16)av;
}

// ---------------- Kernel 2: adj f32 -> f16 ------------------------------------
__global__ void k_cvt(const float* __restrict__ adj, _Float16* __restrict__ adjh) {
  const int i = blockIdx.x * blockDim.x + threadIdx.x;   // N*N = 4M
  adjh[i] = (_Float16)adj[i];
}

// ---------------- Kernel 3: W[b] = adj @ V[b]  (N x 64 per batch) -------------
// 4 waves/block, each wave: 16 rows x 64 cols, K-loop over 2048.
__global__ void k_adjV(const _Float16* __restrict__ adjh, const _Float16* __restrict__ Vt,
                       float* __restrict__ Wf) {
  const int lane = threadIdx.x & 31, w = threadIdx.x >> 5;
  const int b = blockIdx.y;
  const int row0 = blockIdx.x * 64 + w * 16;             // rows of adj / W
  const _Float16* vtb = Vt + (size_t)b * CC * NN;        // Vt[b][c][k], ld = NN
  v8f acc0 = {}, acc1 = {}, acc2 = {}, acc3 = {};
  for (int k0 = 0; k0 < NN; k0 += 32) {
    __builtin_prefetch(adjh + (size_t)(row0 + (lane & 15)) * NN + k0 + 128, 0, 1);
    const v16h a = load_a_tile(adjh, NN, row0, k0, lane);
    acc0 = wmma32(a, load_b_tile(vtb, NN, 0,  k0, lane), acc0);
    acc1 = wmma32(a, load_b_tile(vtb, NN, 16, k0, lane), acc1);
    acc2 = wmma32(a, load_b_tile(vtb, NN, 32, k0, lane), acc2);
    acc3 = wmma32(a, load_b_tile(vtb, NN, 48, k0, lane), acc3);
  }
  const int h = (lane >> 4) & 1, col = lane & 15;
  float* ob = Wf + ((size_t)b * NN + row0) * CC;
#pragma unroll
  for (int g = 0; g < 8; ++g) {
    const int r = g + h * 8;                              // D layout: VGPR g -> row g / g+8
    ob[(size_t)r * CC + col     ] = acc0[g];
    ob[(size_t)r * CC + col + 16] = acc1[g];
    ob[(size_t)r * CC + col + 32] = acc2[g];
    ob[(size_t)r * CC + col + 48] = acc3[g];
  }
}

// ---------------- Kernel 4: column softmax stats over n -----------------------
// colmax[b,m] = max_n S[b,n,m], dinv[b,m] = 1/sum_n exp(S-max); S = Q@K^T.
// Each wave owns 16 m-columns (B operands fixed), streams n with running max.
__global__ void k_stats(const _Float16* __restrict__ Qh, const _Float16* __restrict__ Kh,
                        float* __restrict__ colmax, float* __restrict__ dinv) {
  const int lane = threadIdx.x & 31, w = threadIdx.x >> 5;
  const int b = blockIdx.y;
  const int m0 = blockIdx.x * 64 + w * 16;
  const _Float16* qb = Qh + (size_t)b * NN * CC;
  const _Float16* kb = Kh + (size_t)b * NN * CC;
  const v16h bk0 = load_b_tile(kb, CC, m0, 0, lane);      // K rows m0..m0+15, k 0..31
  const v16h bk1 = load_b_tile(kb, CC, m0, 32, lane);     // k 32..63
  float rmax = -INFINITY, rsum = 0.f;
  for (int n0 = 0; n0 < NN; n0 += 16) {
    v8f s = {};
    s = wmma32(load_a_tile(qb, CC, n0, 0,  lane), bk0, s);
    s = wmma32(load_a_tile(qb, CC, n0, 32, lane), bk1, s);
    float tm = s[0];
#pragma unroll
    for (int g = 1; g < 8; ++g) tm = fmaxf(tm, s[g]);
    tm = fmaxf(tm, __shfl_xor(tm, 16, 32));               // combine rows 0-7 / 8-15 halves
    const float nm = fmaxf(rmax, tm);
    float ts = 0.f;
#pragma unroll
    for (int g = 0; g < 8; ++g) ts += __expf(s[g] - nm);
    ts += __shfl_xor(ts, 16, 32);
    rsum = rsum * __expf(rmax - nm) + ts;                 // online rescale
    rmax = nm;
  }
  if (lane < 16) {
    colmax[(size_t)b * NN + m0 + lane] = rmax;
    dinv  [(size_t)b * NN + m0 + lane] = 1.f / rsum;
  }
}

// ---------------- Kernel 5: Wt[b][c][m] = W[b][m][c] * dinv[b][m]  (f16) ------
__global__ void k_scaleWt(const float* __restrict__ Wf, const float* __restrict__ dinv,
                          _Float16* __restrict__ Wt) {
  const int i = blockIdx.x * blockDim.x + threadIdx.x;    // B*CC*NN, write-coalesced
  const int m = i % NN;
  const int c = (i / NN) % CC;
  const int b = i / (NN * CC);
  Wt[i] = (_Float16)(Wf[((size_t)b * NN + m) * CC + c] * dinv[(size_t)b * NN + m]);
}

// ---------------- Kernel 6: fused  out = exp(Q@K^T - max[m]) @ Wt -------------
// Flash-style: recompute S tiles, exponentiate, LDS-bounce D->A layout, GEMM vs Wt.
__global__ void k_out(const _Float16* __restrict__ Qh, const _Float16* __restrict__ Kh,
                      const _Float16* __restrict__ Wt, const float* __restrict__ colmax,
                      float* __restrict__ out) {
  __shared__ _Float16 pst[4][16][40];                     // per-wave P tile, padded rows
  const int lane = threadIdx.x & 31, w = threadIdx.x >> 5;
  const int b = blockIdx.y;
  const int n0 = blockIdx.x * 64 + w * 16;
  const _Float16* qb  = Qh + (size_t)b * NN * CC;
  const _Float16* kb  = Kh + (size_t)b * NN * CC;
  const _Float16* wtb = Wt + (size_t)b * CC * NN;         // Wt[b][c][m], ld = NN
  const float*    cmb = colmax + (size_t)b * NN;
  const v16h aq0 = load_a_tile(qb, CC, n0, 0, lane);      // Q rows fixed for whole loop
  const v16h aq1 = load_a_tile(qb, CC, n0, 32, lane);
  const int h = (lane >> 4) & 1, lm = lane & 15;
  v8f acc0 = {}, acc1 = {}, acc2 = {}, acc3 = {};
  for (int m0 = 0; m0 < NN; m0 += 32) {
#pragma unroll
    for (int sub = 0; sub < 2; ++sub) {
      const int mm = m0 + sub * 16;
      v8f s = {};
      s = wmma32(aq0, load_b_tile(kb, CC, mm, 0,  lane), s);
      s = wmma32(aq1, load_b_tile(kb, CC, mm, 32, lane), s);
      const float cm = cmb[mm + lm];                      // lane owns column m = mm+lm
#pragma unroll
      for (int g = 0; g < 8; ++g)                         // D layout -> LDS [n][m], f16
        pst[w][g + h * 8][sub * 16 + lm] = (_Float16)__expf(s[g] - cm);
    }
    // same-wave LDS is in-order; fence issue order + wait before readback
    asm volatile("s_wait_dscnt 0" ::: "memory");
    const v8h lo = *(const v8h*)&pst[w][lm][h * 8];       // A-layout readback (transpose)
    const v8h hi = *(const v8h*)&pst[w][lm][16 + h * 8];
    const v16h ap = make_a(lo, hi);
    acc0 = wmma32(ap, load_b_tile(wtb, NN, 0,  m0, lane), acc0);
    acc1 = wmma32(ap, load_b_tile(wtb, NN, 16, m0, lane), acc1);
    acc2 = wmma32(ap, load_b_tile(wtb, NN, 32, m0, lane), acc2);
    acc3 = wmma32(ap, load_b_tile(wtb, NN, 48, m0, lane), acc3);
  }
  float* ob = out + ((size_t)b * NN + n0) * CC;
#pragma unroll
  for (int g = 0; g < 8; ++g) {
    const int r = g + h * 8;
    ob[(size_t)r * CC + lm     ] = acc0[g];
    ob[(size_t)r * CC + lm + 16] = acc1[g];
    ob[(size_t)r * CC + lm + 32] = acc2[g];
    ob[(size_t)r * CC + lm + 48] = acc3[g];
  }
}

extern "C" void kernel_launch(void* const* d_in, const int* in_sizes, int n_in,
                              void* d_out, int out_size, void* d_ws, size_t ws_size,
                              hipStream_t stream) {
  const float* x   = (const float*)d_in[0];
  const float* adj = (const float*)d_in[1];
  const float* Wq  = (const float*)d_in[2];
  const float* bq  = (const float*)d_in[3];
  const float* Wk  = (const float*)d_in[4];
  const float* bk  = (const float*)d_in[5];
  const float* Wv  = (const float*)d_in[6];
  float* out = (float*)d_out;

  // Workspace layout (~32.25 MiB total)
  char* ws = (char*)d_ws;
  _Float16* Qh   = (_Float16*)(ws);                       // B*N*64 f16  = 4 MiB
  _Float16* Kh   = (_Float16*)(ws + (4u  << 20));         // 4 MiB
  _Float16* Vt   = (_Float16*)(ws + (8u  << 20));         // V transposed, 4 MiB
  _Float16* adjh = (_Float16*)(ws + (12u << 20));         // N*N f16 = 8 MiB
  float*    Wf   = (float*)   (ws + (20u << 20));         // adj@V f32, 8 MiB
  _Float16* Wt   = (_Float16*)(ws + (28u << 20));         // scaled+transposed, 4 MiB
  float*    cmax = (float*)   (ws + (32u << 20));         // B*N f32 = 128 KiB
  float*    dinv = (float*)   (ws + (32u << 20) + BB * NN * 4);

  k_qkv    <<<dim3(BB * NN),            dim3(CC),   0, stream>>>(x, Wq, bq, Wk, bk, Wv, Qh, Kh, Vt);
  k_cvt    <<<dim3((NN * NN) / 1024),   dim3(1024), 0, stream>>>(adj, adjh);
  k_adjV   <<<dim3(NN / 64, BB),        dim3(128),  0, stream>>>(adjh, Vt, Wf);
  k_stats  <<<dim3(NN / 64, BB),        dim3(128),  0, stream>>>(Qh, Kh, cmax, dinv);
  k_scaleWt<<<dim3((BB * CC * NN) / 256), dim3(256), 0, stream>>>(Wf, dinv, Wt);
  k_out    <<<dim3(NN / 64, BB),        dim3(128),  0, stream>>>(Qh, Kh, Wt, cmax, out);
  (void)in_sizes; (void)n_in; (void)out_size; (void)ws_size;
}
